// RNN_41059887349766
// MI455X (gfx1250) — matrix operand order repaired and checked by reading, not compiled, not run on hardware
//
#include <hip/hip_runtime.h>

// ---------------------------------------------------------------------------
// GRU LM forward for MI455X (gfx1250): bf16 WMMA GEMMs, TDM-staged LDS weights.
// B=32, S=128, V=32000, E=300 (padded to 320), H=512, 3H=1536.
// ---------------------------------------------------------------------------

typedef __attribute__((ext_vector_type(16))) __bf16 bf16x16;
typedef __attribute__((ext_vector_type(8)))  __bf16 bf16x8;
typedef __attribute__((ext_vector_type(8)))  float  f32x8;
typedef __attribute__((ext_vector_type(4)))  unsigned int u32x4;
typedef __attribute__((ext_vector_type(8)))  int i32x8;
typedef __attribute__((ext_vector_type(4)))  int i32x4;

#define BQ   32      // batch
#define SQ   128     // sequence length
#define TQ   127     // steps = S-1
#define VQ   32000   // vocab
#define EQ   300     // embed dim
#define EP   320     // embed dim padded to mult of 32
#define HQ   512     // hidden
#define GQ   1536    // 3*H
#define MROWS (BQ * TQ)   // 4064 rows of hs / logits

// LDS layout for the GRU kernel (per workgroup, dynamic shared):
#define L_WIH0 0u                 // 3*16*320*2 = 30720
#define L_WHH0 30720u             // 3*16*512*2 = 49152
#define L_WIH1 79872u
#define L_WHH1 129024u
#define L_RED  178176u            // 2 * 32 * 33 floats = 8448 B
#define L_TOTAL (178176u + 8448u) // 186624 B

// ---------------- WMMA helpers ----------------

__device__ inline f32x8 wmma_bf16(bf16x16 a, bf16x16 b, f32x8 c) {
  return __builtin_amdgcn_wmma_f32_16x16x32_bf16(
      /*neg_a=*/false, a, /*neg_b=*/false, b,
      /*c_mod=*/(short)0, c, /*reuse_a=*/false, /*reuse_b=*/false);
}

// A fragment: row-major A[M][lda], 16x32 tile at (mBase,kBase).
__device__ inline bf16x16 load_A(const __bf16* A, int lda, int mBase, int kBase,
                                 int lane) {
  int m  = mBase + (lane & 15);
  int kh = kBase + (lane >> 4) * 16;
  const __bf16* p = A + (size_t)m * lda + kh;
  union { bf16x16 v; bf16x8 h[2]; } u;
  u.h[0] = *(const bf16x8*)p;
  u.h[1] = *(const bf16x8*)(p + 8);
  return u.v;
}

__device__ inline bf16x16 load_A_clamp(const __bf16* A, int lda, int mBase,
                                       int kBase, int lane, int Mmax) {
  int m = mBase + (lane & 15);
  if (m > Mmax - 1) m = Mmax - 1;
  int kh = kBase + (lane >> 4) * 16;
  const __bf16* p = A + (size_t)m * lda + kh;
  union { bf16x16 v; bf16x8 h[2]; } u;
  u.h[0] = *(const bf16x8*)p;
  u.h[1] = *(const bf16x8*)(p + 8);
  return u.v;
}

// B fragment for C = A * W^T with row-major W[N][ldk] in GLOBAL memory.
__device__ inline bf16x16 load_Bt(const __bf16* W, int ldk, int nBase, int kBase,
                                  int lane) {
  int n  = nBase + (lane & 15);
  int kk = kBase + (lane >> 4) * 16;
  return *(const bf16x16*)(W + (size_t)n * ldk + kk);
}

// B fragment from an LDS gate-tile laid out [gate][16 rows][ld].
__device__ inline bf16x16 load_Bt_lds(const __bf16* tile, int ld, int gate,
                                      int kBase, int lane) {
  int n  = gate * 16 + (lane & 15);
  int kk = kBase + (lane >> 4) * 16;
  return *(const bf16x16*)(tile + n * ld + kk);
}

__device__ inline float sigmoidf_(float x) { return 1.0f / (1.0f + expf(-x)); }

// ---------------- device-wide barrier (32 resident workgroups) ----------------

__device__ inline void grid_barrier(int* bar, int nwg) {
  __threadfence();
  __syncthreads();
  if (threadIdx.x == 0) {
    int gen = __hip_atomic_load(&bar[1], __ATOMIC_ACQUIRE, __HIP_MEMORY_SCOPE_AGENT);
    int prev = __hip_atomic_fetch_add(&bar[0], 1, __ATOMIC_ACQ_REL,
                                      __HIP_MEMORY_SCOPE_AGENT);
    if (prev == nwg - 1) {
      __hip_atomic_store(&bar[0], 0, __ATOMIC_RELAXED, __HIP_MEMORY_SCOPE_AGENT);
      __hip_atomic_fetch_add(&bar[1], 1, __ATOMIC_RELEASE,
                             __HIP_MEMORY_SCOPE_AGENT);
    } else {
      while (__hip_atomic_load(&bar[1], __ATOMIC_ACQUIRE,
                               __HIP_MEMORY_SCOPE_AGENT) == gen) {
        __builtin_amdgcn_s_sleep(4);
      }
    }
  }
  __syncthreads();
  __threadfence();
}

// ---------------- TDM: stage one weight matrix's 3 gate tiles into LDS -------
// 3-D tile: dim0 = ld elems (one row), dim1 = 16 rows, dim2 = 3 gate planes
// separated by 512 rows (tensor_dim1_stride = 512*ld). bf16 => data_size = 2B.
// This toolchain's builtin is the 6-arg form:
//   (u32x4 g0, i32x8 g1, i32x4 g2, i32x4 g3, i32x8 g4_unused, i32 cpol)

#if __has_builtin(__builtin_amdgcn_tensor_load_to_lds)
#define HAVE_TDM 1
__device__ inline void tdm_load_gates(unsigned lds_off, const __bf16* gsrc,
                                      int ld) {
  unsigned long long ga = (unsigned long long)(const void*)gsrc;
  long long s1 = (long long)ld * 512;   // plane stride (elements)
  u32x4 g0;
  g0[0] = 1u;                                       // count=1 (valid), user mode
  g0[1] = lds_off;                                  // LDS byte address
  g0[2] = (unsigned)(ga & 0xFFFFFFFFu);             // global_addr[31:0]
  g0[3] = (unsigned)((ga >> 32) & 0x01FFFFFFu) | (2u << 30);  // addr[56:32]|type=2
  i32x8 g1;
  g1[0] = 1 << 16;                                  // data_size=1 (2 bytes)
  g1[1] = (ld & 0xFFFF) << 16;                      // tensor_dim0[15:0]
  g1[2] = (ld >> 16) | (16 << 16);                  // td0 hi | tensor_dim1=16 lo
  g1[3] = (ld << 16);                               // td1 hi(0) | tile_dim0=ld
  g1[4] = 16 | (3 << 16);                           // tile_dim1=16 | tile_dim2=3
  g1[5] = ld;                                       // tensor_dim0_stride lo32
  g1[6] = ((int)(s1 & 0xFFFF)) << 16;               // td0s hi(0) | td1_stride lo16
  g1[7] = (int)(s1 >> 16);                          // tensor_dim1_stride[47:16]
  i32x4 g2;
  g2[0] = 3;                                        // tensor_dim2 = 3
  g2[1] = 0;                                        // tensor_dim3 (unused)
  g2[2] = 0;                                        // tensor_dim2_stride lo
  g2[3] = 0;                                        // stride hi | tile_dim3=0
  i32x4 g3;
  g3[0] = 0; g3[1] = 0; g3[2] = 0; g3[3] = 0;
  i32x8 g4;
#pragma unroll
  for (int i = 0; i < 8; ++i) g4[i] = 0;            // VADDR4-style group: unused
  __builtin_amdgcn_tensor_load_to_lds(g0, g1, g2, g3, g4, 0);
}
#else
#define HAVE_TDM 0
#endif

// Fallback / portable cooperative copy of one gate block (16 rows x ld).
__device__ inline void stage_copy(__bf16* dst, const __bf16* src, int nelem) {
  for (int i = threadIdx.x * 8; i < nelem; i += blockDim.x * 8)
    *(bf16x8*)(dst + i) = *(const bf16x8*)(src + i);
}

// ---------------- setup kernels ----------------

__global__ void init_state_kernel(int* bar, __bf16* h0buf, __bf16* h1buf) {
  int i = blockIdx.x * blockDim.x + threadIdx.x;
  if (i < 4) bar[i] = 0;
  for (int j = i; j < 2 * BQ * HQ; j += gridDim.x * blockDim.x) {
    h0buf[j] = (__bf16)0.0f;
    h1buf[j] = (__bf16)0.0f;
  }
}

__global__ void embed_kernel(const int* sent, const float* wordvec, __bf16* xbf) {
  int tb = blockIdx.x;           // 0 .. 127*32-1
  int t = tb >> 5;
  int b = tb & 31;
  int word = sent[b * SQ + t];
  const float* src = wordvec + (size_t)word * EQ;
  __bf16* dst = xbf + ((size_t)t * BQ + b) * EP;
  for (int e = threadIdx.x; e < EP; e += blockDim.x)
    dst[e] = (e < EQ) ? (__bf16)src[e] : (__bf16)0.0f;
}

__global__ void convert_pad_kernel(const float* src, __bf16* dst, int rows,
                                   int ksrc, int kdst) {
  int total = rows * kdst;
  for (int i = blockIdx.x * blockDim.x + threadIdx.x; i < total;
       i += gridDim.x * blockDim.x) {
    int row = i / kdst;
    int k = i - row * kdst;
    dst[i] = (k < ksrc) ? (__bf16)src[(size_t)row * ksrc + k] : (__bf16)0.0f;
  }
}

// ---------------- GRU recurrence ----------------
// 32 WGs x 4 waves. WG owns N-tile nt = blockIdx.x (16 hidden columns).
// Wave w: mt = w&1 (batch half), kh = w>>1 (K half). K-split partials are
// combined through LDS; weights live in LDS (staged once by TDM).

__device__ inline void gru_accum(const __bf16* X, int ldx, int k0, int k1,
                                 const __bf16* Wlds, int ld, int lane, int mBase,
                                 f32x8& a_r, f32x8& a_z, f32x8& a_n) {
  for (int k = k0; k < k1; k += 32) {
    bf16x16 a = load_A(X, ldx, mBase, k, lane);
    a_r = wmma_bf16(a, load_Bt_lds(Wlds, ld, 0, k, lane), a_r);
    a_z = wmma_bf16(a, load_Bt_lds(Wlds, ld, 1, k, lane), a_z);
    a_n = wmma_bf16(a, load_Bt_lds(Wlds, ld, 2, k, lane), a_n);
  }
}

__device__ inline void gru_layer_split(
    const __bf16* X, int ldx, int Kx, const __bf16* WihL, const __bf16* WhhL,
    const float* bih, const float* bhh, const __bf16* Hprev, __bf16* Hnext,
    __bf16* hs, int t, int mBase, int nt, int kh, int lane, float* red) {
  f32x8 acc_r = {}, acc_z = {}, acc_in = {}, acc_hn = {};
  int kx0 = kh * (Kx / 2);
  gru_accum(X, ldx, kx0, kx0 + Kx / 2, WihL, ldx, lane, mBase,
            acc_r, acc_z, acc_in);
  int kb0 = kh * (HQ / 2);
  gru_accum(Hprev, HQ, kb0, kb0 + HQ / 2, WhhL, HQ, lane, mBase,
            acc_r, acc_z, acc_hn);

  float* slot = red + (mBase ? 32 * 33 : 0) + lane * 33;  // padded: no conflicts
  if (kh == 1) {
#pragma unroll
    for (int r = 0; r < 8; ++r) {
      slot[r]      = acc_r[r];
      slot[8 + r]  = acc_z[r];
      slot[16 + r] = acc_in[r];
      slot[24 + r] = acc_hn[r];
    }
  }
  __syncthreads();
  if (kh == 0) {
    const int jn  = nt * 16 + (lane & 15);
    const float br  = bih[jn] + bhh[jn];
    const float bz  = bih[HQ + jn] + bhh[HQ + jn];
    const float bin = bih[2 * HQ + jn];
    const float bhn = bhh[2 * HQ + jn];
    const int mOff = (lane >> 4) * 8;
#pragma unroll
    for (int r = 0; r < 8; ++r) {
      int m = mBase + mOff + r;
      float rr = sigmoidf_(acc_r[r] + slot[r] + br);
      float zz = sigmoidf_(acc_z[r] + slot[8 + r] + bz);
      float nn = tanhf(acc_in[r] + slot[16 + r] + bin +
                       rr * (acc_hn[r] + slot[24 + r] + bhn));
      float hp = (float)Hprev[m * HQ + jn];
      float hnew = (1.0f - zz) * nn + zz * hp;
      __bf16 hb = (__bf16)hnew;
      Hnext[m * HQ + jn] = hb;
      if (hs) hs[((size_t)m * TQ + t) * HQ + jn] = hb;
    }
  }
}

__global__ __launch_bounds__(128) void gru_persistent_kernel(
    const __bf16* xbf, const __bf16* Wih0b, const __bf16* Whh0b,
    const __bf16* Wih1b, const __bf16* Whh1b, const float* bih0,
    const float* bhh0, const float* bih1, const float* bhh1, __bf16* h0buf,
    __bf16* h1buf, __bf16* hs, int* bar) {
  extern __shared__ char smem[];
  __bf16* wih0 = (__bf16*)(smem + L_WIH0);
  __bf16* whh0 = (__bf16*)(smem + L_WHH0);
  __bf16* wih1 = (__bf16*)(smem + L_WIH1);
  __bf16* whh1 = (__bf16*)(smem + L_WHH1);
  float*  red  = (float*)(smem + L_RED);

  const int lane = threadIdx.x & 31;
  const int wv   = threadIdx.x >> 5;
  const int mBase = (wv & 1) * 16;
  const int kh    = wv >> 1;
  const int nt    = blockIdx.x;      // 0..31
  const int nwg   = gridDim.x;

  // ---- stage this WG's weight working set (178 KB) into LDS ----
#if HAVE_TDM
  if (wv == 0) {
    tdm_load_gates(L_WIH0, Wih0b + (size_t)(nt * 16) * EP, EP);
    tdm_load_gates(L_WHH0, Whh0b + (size_t)(nt * 16) * HQ, HQ);
    tdm_load_gates(L_WIH1, Wih1b + (size_t)(nt * 16) * HQ, HQ);
    tdm_load_gates(L_WHH1, Whh1b + (size_t)(nt * 16) * HQ, HQ);
    __builtin_amdgcn_s_wait_tensorcnt(0);
  }
#else
  for (int g = 0; g < 3; ++g) {
    stage_copy(wih0 + g * 16 * EP, Wih0b + (size_t)(g * HQ + nt * 16) * EP, 16 * EP);
    stage_copy(whh0 + g * 16 * HQ, Whh0b + (size_t)(g * HQ + nt * 16) * HQ, 16 * HQ);
    stage_copy(wih1 + g * 16 * HQ, Wih1b + (size_t)(g * HQ + nt * 16) * HQ, 16 * HQ);
    stage_copy(whh1 + g * 16 * HQ, Whh1b + (size_t)(g * HQ + nt * 16) * HQ, 16 * HQ);
  }
#endif
  __syncthreads();

  int p = 0;
#pragma unroll 1
  for (int t = 0; t < TQ; ++t) {
    const __bf16* X = xbf + (size_t)t * BQ * EP;
    gru_layer_split(X, EP, EP, wih0, whh0, bih0, bhh0,
                    h0buf + p * BQ * HQ, h0buf + (1 - p) * BQ * HQ,
                    nullptr, 0, mBase, nt, kh, lane, red);
    grid_barrier(bar, nwg);
    gru_layer_split(h0buf + (1 - p) * BQ * HQ, HQ, HQ, wih1, whh1, bih1, bhh1,
                    h1buf + p * BQ * HQ, h1buf + (1 - p) * BQ * HQ,
                    hs, t, mBase, nt, kh, lane, red);
    grid_barrier(bar, nwg);
    p ^= 1;
  }
}

// ---------------- output projection: logits = hs @ Wout^T + bout -------------

__global__ __launch_bounds__(256) void logits_kernel(const __bf16* A,
                                                     const __bf16* W,
                                                     const float* bout,
                                                     float* C) {
  const int lane = threadIdx.x & 31;
  const int wv = threadIdx.x >> 5;
  const int wm = wv >> 2;
  const int wn = wv & 3;
  const int mWave = blockIdx.x * 64 + wm * 32;
  const int nWave = blockIdx.y * 256 + wn * 64;

  f32x8 acc[2][4];
#pragma unroll
  for (int i = 0; i < 2; ++i)
#pragma unroll
    for (int j = 0; j < 4; ++j) acc[i][j] = (f32x8){};

  for (int k = 0; k < HQ; k += 32) {
    bf16x16 a0 = load_A_clamp(A, HQ, mWave,      k, lane, MROWS);
    bf16x16 a1 = load_A_clamp(A, HQ, mWave + 16, k, lane, MROWS);
#pragma unroll
    for (int j = 0; j < 4; ++j) {
      bf16x16 b = load_Bt(W, HQ, nWave + j * 16, k, lane);
      acc[0][j] = wmma_bf16(a0, b, acc[0][j]);
      acc[1][j] = wmma_bf16(a1, b, acc[1][j]);
    }
  }

  float bb[4];
#pragma unroll
  for (int j = 0; j < 4; ++j) bb[j] = bout[nWave + j * 16 + (lane & 15)];
  const int mOff = (lane >> 4) * 8;
#pragma unroll
  for (int i = 0; i < 2; ++i) {
#pragma unroll
    for (int r = 0; r < 8; ++r) {
      int m = mWave + i * 16 + mOff + r;
      if (m < MROWS) {
#pragma unroll
        for (int j = 0; j < 4; ++j) {
          int col = nWave + j * 16 + (lane & 15);
          C[(size_t)m * VQ + col] = acc[i][j][r] + bb[j];
        }
      }
    }
  }
}

// ---------------- log-softmax NLL per row ----------------

__global__ __launch_bounds__(256) void nll_kernel(const float* logits,
                                                  const int* sent, float* nll) {
  const int m = blockIdx.x;
  const int b = m / TQ;
  const int t = m - b * TQ;
  const float* row = logits + (size_t)m * VQ;
  __shared__ float red[256];

  float mx = -3.4e38f;
  for (int v = threadIdx.x; v < VQ; v += 256) mx = fmaxf(mx, row[v]);
  red[threadIdx.x] = mx;
  __syncthreads();
  for (int s = 128; s > 0; s >>= 1) {
    if (threadIdx.x < s)
      red[threadIdx.x] = fmaxf(red[threadIdx.x], red[threadIdx.x + s]);
    __syncthreads();
  }
  mx = red[0];
  __syncthreads();

  float sum = 0.0f;
  for (int v = threadIdx.x; v < VQ; v += 256) sum += expf(row[v] - mx);
  red[threadIdx.x] = sum;
  __syncthreads();
  for (int s = 128; s > 0; s >>= 1) {
    if (threadIdx.x < s) red[threadIdx.x] += red[threadIdx.x + s];
    __syncthreads();
  }

  if (threadIdx.x == 0) {
    int tgt = sent[b * SQ + t + 1];
    nll[m] = -(row[tgt] - mx - logf(red[0]));
  }
}

// ---------------- final masked-mean loss ----------------

__global__ __launch_bounds__(64) void loss_kernel(const float* nll,
                                                  const int* length,
                                                  float* out) {
  __shared__ float ps[BQ];
  int b = threadIdx.x;
  if (b < BQ) {
    int len = length[b];
    int nsteps = len - 1;
    if (nsteps > TQ) nsteps = TQ;
    if (nsteps < 0) nsteps = 0;
    float s = 0.0f;
    for (int t = 0; t < nsteps; ++t) s += nll[b * TQ + t];
    float denom = fmaxf((float)(len - 1), 1.0f);
    float valid = (len > 1) ? 1.0f : 0.0f;
    ps[b] = (s / denom) * valid;
  }
  __syncthreads();
  if (threadIdx.x == 0) {
    float tot = 0.0f, nv = 0.0f;
    for (int i = 0; i < BQ; ++i) {
      tot += ps[i];
      nv += (length[i] > 1) ? 1.0f : 0.0f;
    }
    out[0] = tot / fmaxf(nv, 1.0f);
  }
}

// ---------------- host: workspace layout + launch sequence ----------------

extern "C" void kernel_launch(void* const* d_in, const int* in_sizes, int n_in,
                              void* d_out, int out_size, void* d_ws,
                              size_t ws_size, hipStream_t stream) {
  (void)in_sizes; (void)n_in; (void)out_size; (void)ws_size;

  const int*   sent    = (const int*)d_in[0];
  const int*   length  = (const int*)d_in[1];
  const float* wordvec = (const float*)d_in[2];
  const float* Wih0    = (const float*)d_in[3];
  const float* Whh0    = (const float*)d_in[4];
  const float* bih0    = (const float*)d_in[5];
  const float* bhh0    = (const float*)d_in[6];
  const float* Wih1    = (const float*)d_in[7];
  const float* Whh1    = (const float*)d_in[8];
  const float* bih1    = (const float*)d_in[9];
  const float* bhh1    = (const float*)d_in[10];
  const float* Wout    = (const float*)d_in[11];
  const float* bout    = (const float*)d_in[12];

  char* ws = (char*)d_ws;
  size_t off = 0;
  auto take = [&](size_t bytes) {
    char* p = ws + off;
    off = (off + bytes + 255) & ~(size_t)255;
    return p;
  };
  int*    bar    = (int*)take(16);
  __bf16* xbf    = (__bf16*)take((size_t)TQ * BQ * EP * 2);
  __bf16* Wih0b  = (__bf16*)take((size_t)GQ * EP * 2);
  __bf16* Whh0b  = (__bf16*)take((size_t)GQ * HQ * 2);
  __bf16* Wih1b  = (__bf16*)take((size_t)GQ * HQ * 2);
  __bf16* Whh1b  = (__bf16*)take((size_t)GQ * HQ * 2);
  __bf16* Woutb  = (__bf16*)take((size_t)VQ * HQ * 2);
  __bf16* h0buf  = (__bf16*)take((size_t)2 * BQ * HQ * 2);
  __bf16* h1buf  = (__bf16*)take((size_t)2 * BQ * HQ * 2);
  __bf16* hsb    = (__bf16*)take((size_t)MROWS * HQ * 2);
  float*  nllbuf = (float*)take((size_t)MROWS * 4);

  float* out    = (float*)d_out;
  float* logits = out + 1;   // (B, S-1, V) flat == row m = b*127+t

  init_state_kernel<<<64, 256, 0, stream>>>(bar, h0buf, h1buf);
  embed_kernel<<<TQ * BQ, 64, 0, stream>>>(sent, wordvec, xbf);
  convert_pad_kernel<<<1024, 256, 0, stream>>>(Wih0, Wih0b, GQ, EQ, EP);
  convert_pad_kernel<<<1024, 256, 0, stream>>>(Whh0, Whh0b, GQ, HQ, HQ);
  convert_pad_kernel<<<1024, 256, 0, stream>>>(Wih1, Wih1b, GQ, HQ, HQ);
  convert_pad_kernel<<<1024, 256, 0, stream>>>(Whh1, Whh1b, GQ, HQ, HQ);
  convert_pad_kernel<<<4096, 256, 0, stream>>>(Wout, Woutb, VQ, HQ, HQ);

  gru_persistent_kernel<<<32, 128, L_TOTAL, stream>>>(
      xbf, Wih0b, Whh0b, Wih1b, Whh1b, bih0, bhh0, bih1, bhh1, h0buf, h1buf,
      hsb, bar);

  dim3 ggrid(64, 125);  // 64*64=4096 >= 4064 rows ; 125*256 = 32000 cols
  logits_kernel<<<ggrid, 256, 0, stream>>>(hsb, Woutb, bout, logits);

  nll_kernel<<<MROWS, 256, 0, stream>>>(logits, sent, nllbuf);
  loss_kernel<<<1, 64, 0, stream>>>(nllbuf, length, out);
}
